// pNN_39599598469769
// MI455X (gfx1250) — compile-verified, matrix operand order checked by reading, not run on hardware
//
#include <hip/hip_runtime.h>
#include <math.h>

#define GMIN_C  0.01f
#define GMAX_C  10.0f
#define PGMIN_C 1e-6f

// Flip to 0 if the async-to-LDS inline asm is rejected by the assembler.
#ifndef USE_ASYNC_LDS
#define USE_ASYNC_LDS 1
#endif

typedef __attribute__((ext_vector_type(16))) __bf16       v16bf;
typedef __attribute__((ext_vector_type(8)))  float        v8f;
typedef __attribute__((ext_vector_type(4)))  unsigned int u32x4;

__device__ __forceinline__ float act_fn(float z) { return tanhf((z - 0.1f) * 5.0f); }
__device__ __forceinline__ float inv_fn(float x) { return -tanhf(5.0f * x); }

#if USE_ASYNC_LDS
// GLOBAL_LOAD_ASYNC_TO_LDS_B128: vdst = LDS byte offset (VGPR), vaddr = 64-bit
// global address, GV mode (saddr = off). Tracked by ASYNCcnt.
__device__ __forceinline__ void async_ld_b128(unsigned lds_off, const void* gaddr) {
  asm volatile("global_load_async_to_lds_b128 %0, %1, off"
               :: "v"(lds_off), "v"(gaddr) : "memory");
}
__device__ __forceinline__ void wait_async0() {
  asm volatile("s_wait_asynccnt 0x0" ::: "memory");
}
#endif

// ---------------------------------------------------------------------------
__global__ void zero_pow_kernel(float* p) { *p = 0.0f; }

// ---------------------------------------------------------------------------
// Build bf16 A-panel [E x Kpad] = [a_ext | a_neg | 0-pad], plus per-column
// Sxx[m] = sum_e a_ext^2, Snn[m] = sum_e a_neg^2.
__global__ void act_prep_kernel(const float* __restrict__ act, int E, int Min, int M,
                                int Kpad, __bf16* __restrict__ Ap,
                                float* __restrict__ Sxx, float* __restrict__ Snn)
{
  int t  = threadIdx.x;
  int c  = t & 63, r0 = t >> 6;
  int m  = blockIdx.x * 64 + c;
  bool valid = (m < M);
  float cst = (m == Min) ? 1.0f : 0.0f;   // bias column = 1, last column = 0
  float sx = 0.f, sn = 0.f;
  for (int e = r0; e < E; e += 4) {
    float a  = (m < Min) ? act[(size_t)e * Min + m] : cst;
    float an = inv_fn(a);
    if (valid) {
      size_t base = (size_t)e * Kpad;
      Ap[base + m]     = (__bf16)a;
      Ap[base + M + m] = (__bf16)an;
      sx += a * a;
      sn += an * an;
    }
  }
  __shared__ float red[2][4][64];
  red[0][r0][c] = sx;
  red[1][r0][c] = sn;
  __syncthreads();
  if (r0 == 0 && valid) {
    Sxx[m] = red[0][0][c] + red[0][1][c] + red[0][2][c] + red[0][3][c];
    Snn[m] = red[1][0][c] + red[1][1][c] + red[1][2][c] + red[1][3][c];
  }
}

// ---------------------------------------------------------------------------
// Per-column stats of theta: s[n], gscale[n] = PGMIN/min|theta_raw|, Gcol[n],
// and power terms sum_m g*(pos?Sxx:Snn) accumulated into powAcc.
__global__ void theta_stats_kernel(const float* __restrict__ theta, int M, int N,
                                   const float* __restrict__ Sxx, const float* __restrict__ Snn,
                                   float* __restrict__ s_arr, float* __restrict__ gsc,
                                   float* __restrict__ Gcol, float* __restrict__ powAcc,
                                   float invE)
{
  int t = threadIdx.x;
  int c = t & 63, r0 = t >> 6;
  int n = blockIdx.x * 64 + c;
  __shared__ float rs[2][4][64];
  __shared__ float gsh[64];
  __shared__ float psum;
  if (t == 0) psum = 0.f;

  float s = 0.f, mn = 3.402823e38f;
  for (int m = r0; m < M; m += 4) {
    float th = theta[(size_t)m * N + n];
    float tc = fminf(fmaxf(th, -GMAX_C), GMAX_C);
    float ta = fabsf(tc);
    if (ta < GMIN_C) ta = 0.f;
    s += ta;
    mn = fminf(mn, fabsf(th));
  }
  rs[0][r0][c] = s;
  rs[1][r0][c] = mn;
  __syncthreads();
  if (r0 == 0) {
    float S  = rs[0][0][c] + rs[0][1][c] + rs[0][2][c] + rs[0][3][c];
    float MN = fminf(fminf(rs[1][0][c], rs[1][1][c]), fminf(rs[1][2][c], rs[1][3][c]));
    float gs = PGMIN_C / MN;
    s_arr[n] = S;
    gsc[n]   = gs;
    gsh[c]   = gs;
  }
  __syncthreads();
  float gs = gsh[c];
  float gc = 0.f, t13 = 0.f;
  for (int m = r0; m < M; m += 4) {
    float th = theta[(size_t)m * N + n];
    float tc = fminf(fmaxf(th, -GMAX_C), GMAX_C);
    float te = (fabsf(tc) < GMIN_C) ? 0.f : tc;
    float g  = fabsf(th) * gs;
    gc  += g;
    t13 += g * ((te >= 0.f) ? Sxx[m] : Snn[m]);
  }
  atomicAdd(&psum, t13);
  rs[0][r0][c] = gc;
  __syncthreads();
  if (r0 == 0)
    Gcol[n] = rs[0][0][c] + rs[0][1][c] + rs[0][2][c] + rs[0][3][c];
  if (t == 0) atomicAdd(powAcc, psum * invE);
}

// ---------------------------------------------------------------------------
// Emit bf16 B matrix, n-major [2N x Kpad]:
//   row n     : k in [0,M) -> W*pos, k in [M,2M) -> W*neg     (z part)
//   row N + n : k in [0,M) -> g*pos, k in [M,2M) -> g*neg     (power part)
__global__ void theta_emit_kernel(const float* __restrict__ theta, int M, int N, int Kpad,
                                  const float* __restrict__ s_arr, const float* __restrict__ gsc,
                                  __bf16* __restrict__ B)
{
  __shared__ __bf16 tile[4][64][65];
  int t  = threadIdx.x;
  int nl = t & 63, rr = t >> 6;
  int n  = blockIdx.y * 64 + nl;
  int m0 = blockIdx.x * 64;
  float S    = s_arr[n];
  float srec = (S > 0.f) ? (1.0f / S) : 0.f;
  float gs   = gsc[n];
#pragma unroll 4
  for (int i = 0; i < 16; ++i) {
    int ml = rr + i * 4;
    int m  = m0 + ml;
    float th = (m < M) ? theta[(size_t)m * N + n] : 0.f;
    float tc = fminf(fmaxf(th, -GMAX_C), GMAX_C);
    float te = (fabsf(tc) < GMIN_C) ? 0.f : tc;
    float W  = fabsf(te) * srec;
    float g  = fabsf(th) * gs;
    bool pos = (te >= 0.f);
    tile[0][ml][nl] = (__bf16)(pos ? W : 0.f);
    tile[1][ml][nl] = (__bf16)(pos ? 0.f : W);
    tile[2][ml][nl] = (__bf16)(pos ? g : 0.f);
    tile[3][ml][nl] = (__bf16)(pos ? 0.f : g);
  }
  __syncthreads();
#pragma unroll 4
  for (int i = 0; i < 16; ++i) {
    int nl2 = rr + i * 4;
    int ml2 = t & 63;
    int m   = m0 + ml2;
    int nn  = blockIdx.y * 64 + nl2;
    if (m < M) {
      size_t rz = (size_t)nn * Kpad;
      size_t rp = (size_t)(N + nn) * Kpad;
      B[rz + m]     = tile[0][ml2][nl2];
      B[rz + M + m] = tile[1][ml2][nl2];
      B[rp + m]     = tile[2][ml2][nl2];
      B[rp + M + m] = tile[3][ml2][nl2];
    }
  }
}

// ---------------------------------------------------------------------------
// Zero the K-padding columns [2M, Kpad) of A-panel and B.
__global__ void pad_zero_kernel(__bf16* __restrict__ Ap, __bf16* __restrict__ B,
                                int E, int twoN, int twoM, int Kpad)
{
  int padc = Kpad - twoM;
  long long idx  = (long long)blockIdx.x * blockDim.x + threadIdx.x;
  long long totA = (long long)E * padc;
  long long totB = (long long)twoN * padc;
  if (idx < totA) {
    long long e = idx / padc, k = twoM + idx % padc;
    Ap[e * (long long)Kpad + k] = (__bf16)0.f;
  } else if (idx < totA + totB) {
    long long j = idx - totA;
    long long nr = j / padc, k = twoM + j % padc;
    B[nr * (long long)Kpad + k] = (__bf16)0.f;
  }
}

// ---------------------------------------------------------------------------
// Fused WMMA GEMM: per 128x128 tile computes z = A@Bz and uv = A@Bp with
// v_wmma_f32_16x16x32_bf16, writes act(z), accumulates power partials.
//
// LDS row = 32 K-halves permuted [K0-7, K16-23 | K8-15, K24-31] so each lane's
// 16-half WMMA operand is one contiguous 32B LDS load; the two 32B chunks of a
// row are XOR-swizzled by row bit2 (applied on both fill and read sides) to
// stagger banks. Double-buffered LDS, one barrier per K-step; tiles streamed
// via GLOBAL_LOAD_ASYNC_TO_LDS_B128 (ASYNCcnt) when USE_ASYNC_LDS.
__global__ __launch_bounds__(256) void gemm_fused_kernel(
    const __bf16* __restrict__ A,     // [E x Kpad], row-major
    const __bf16* __restrict__ B,     // [2N x Kpad], n-major (k contiguous)
    const float* __restrict__ Gcol,   // [N]
    float* __restrict__ actOut,       // [E x N]
    float* __restrict__ powAcc,
    int E, int N, int Kpad, float invE)
{
  constexpr int BK  = 32;
  constexpr int STR = 32;             // halves per LDS row (64B, 32B aligned)
  __shared__ __bf16 As [2][128 * STR];
  __shared__ __bf16 Bzs[2][128 * STR];
  __shared__ __bf16 Bps[2][128 * STR];
  __shared__ float psum;

  int t    = threadIdx.x;
  int lane = t & 31;
  int wave = t >> 5;
  int wr   = wave >> 1;               // 0..3 : 32-row group
  int wc   = wave & 1;                // 0..1 : 64-col group

  int eBase = blockIdx.x * 128;
  int nBase = blockIdx.y * 128;

  // fill mapping: 256 threads cover 128 rows x 2 K-segments of 16 halves
  int lrow = t >> 1;
  int seg  = t & 1;
  unsigned sw = (unsigned)(((lrow >> 2) & 1) << 4);     // chunk swizzle (halves)
  unsigned o0 = (unsigned)(lrow * STR) + (((unsigned)(seg ? 8  : 0)) ^ sw);
  unsigned o1 = (unsigned)(lrow * STR) + (((unsigned)(seg ? 24 : 16)) ^ sw);

  const __bf16* Aptr  = A + (size_t)(eBase + lrow) * Kpad + seg * 16;
  const __bf16* Bzptr = B + (size_t)(nBase + lrow) * Kpad + seg * 16;
  const __bf16* Bpptr = B + (size_t)(N + nBase + lrow) * Kpad + seg * 16;

  if (t == 0) psum = 0.f;

  v8f accZ[2][4], accP[2][4];
#pragma unroll
  for (int i = 0; i < 2; ++i)
#pragma unroll
    for (int j = 0; j < 4; ++j) {
      accZ[i][j] = (v8f){0.f, 0.f, 0.f, 0.f, 0.f, 0.f, 0.f, 0.f};
      accP[i][j] = (v8f){0.f, 0.f, 0.f, 0.f, 0.f, 0.f, 0.f, 0.f};
    }

  int lr    = lane & 15;
  int kbase = (lane < 16) ? 0 : 16;

#if USE_ASYNC_LDS
  // prologue: stream tile 0 into buffer 0
  async_ld_b128((unsigned)(uintptr_t)&As [0][o0], Aptr);
  async_ld_b128((unsigned)(uintptr_t)&As [0][o1], Aptr + 8);
  async_ld_b128((unsigned)(uintptr_t)&Bzs[0][o0], Bzptr);
  async_ld_b128((unsigned)(uintptr_t)&Bzs[0][o1], Bzptr + 8);
  async_ld_b128((unsigned)(uintptr_t)&Bps[0][o0], Bpptr);
  async_ld_b128((unsigned)(uintptr_t)&Bps[0][o1], Bpptr + 8);
  Aptr += BK; Bzptr += BK; Bpptr += BK;
#else
  {
    u32x4 a0 = *(const u32x4*)(Aptr);
    u32x4 a1 = *(const u32x4*)(Aptr + 8);
    u32x4 b0 = *(const u32x4*)(Bzptr);
    u32x4 b1 = *(const u32x4*)(Bzptr + 8);
    u32x4 c0 = *(const u32x4*)(Bpptr);
    u32x4 c1 = *(const u32x4*)(Bpptr + 8);
    Aptr += BK; Bzptr += BK; Bpptr += BK;
    *(u32x4*)&As [0][o0] = a0;  *(u32x4*)&As [0][o1] = a1;
    *(u32x4*)&Bzs[0][o0] = b0;  *(u32x4*)&Bzs[0][o1] = b1;
    *(u32x4*)&Bps[0][o0] = c0;  *(u32x4*)&Bps[0][o1] = c1;
  }
#endif

  int cur = 0;
  for (int kb = 0; kb < Kpad; kb += BK) {
    bool more = (kb + BK) < Kpad;
#if USE_ASYNC_LDS
    wait_async0();                    // my async fills of buf[cur] landed
    __syncthreads();                  // everyone's fills landed
    if (more) {
      int nxt = cur ^ 1;
      async_ld_b128((unsigned)(uintptr_t)&As [nxt][o0], Aptr);
      async_ld_b128((unsigned)(uintptr_t)&As [nxt][o1], Aptr + 8);
      async_ld_b128((unsigned)(uintptr_t)&Bzs[nxt][o0], Bzptr);
      async_ld_b128((unsigned)(uintptr_t)&Bzs[nxt][o1], Bzptr + 8);
      async_ld_b128((unsigned)(uintptr_t)&Bps[nxt][o0], Bpptr);
      async_ld_b128((unsigned)(uintptr_t)&Bps[nxt][o1], Bpptr + 8);
      Aptr += BK; Bzptr += BK; Bpptr += BK;
    }
#else
    __syncthreads();                  // buf[cur] stores visible
    u32x4 ra0, ra1, rb0, rb1, rc0, rc1;
    if (more) {
      ra0 = *(const u32x4*)(Aptr);
      ra1 = *(const u32x4*)(Aptr + 8);
      rb0 = *(const u32x4*)(Bzptr);
      rb1 = *(const u32x4*)(Bzptr + 8);
      rc0 = *(const u32x4*)(Bpptr);
      rc1 = *(const u32x4*)(Bpptr + 8);
      __builtin_prefetch(Aptr  + 2 * BK, 0, 0);
      __builtin_prefetch(Bzptr + 2 * BK, 0, 0);
      __builtin_prefetch(Bpptr + 2 * BK, 0, 0);
      Aptr += BK; Bzptr += BK; Bpptr += BK;
    }
#endif

    // Preload ALL fragments, then issue all 16 WMMAs back-to-back.
    v16bf af[2], bzf[4], bpf[4];
#pragma unroll
    for (int fr = 0; fr < 2; ++fr) {
      int row = wr * 32 + fr * 16 + lr;
      int ko  = kbase ^ (((row >> 2) & 1) << 4);
      af[fr] = *(const v16bf*)&As[cur][row * STR + ko];
    }
#pragma unroll
    for (int fc = 0; fc < 4; ++fc) {
      int row = wc * 64 + fc * 16 + lr;
      int ko  = kbase ^ (((row >> 2) & 1) << 4);
      bzf[fc] = *(const v16bf*)&Bzs[cur][row * STR + ko];
      bpf[fc] = *(const v16bf*)&Bps[cur][row * STR + ko];
    }
#pragma unroll
    for (int fc = 0; fc < 4; ++fc)
#pragma unroll
      for (int fr = 0; fr < 2; ++fr)
        accZ[fr][fc] = __builtin_amdgcn_wmma_f32_16x16x32_bf16(
            false, af[fr], false, bzf[fc], (short)0, accZ[fr][fc], false, false);
#pragma unroll
    for (int fc = 0; fc < 4; ++fc)
#pragma unroll
      for (int fr = 0; fr < 2; ++fr)
        accP[fr][fc] = __builtin_amdgcn_wmma_f32_16x16x32_bf16(
            false, af[fr], false, bpf[fc], (short)0, accP[fr][fc], false, false);

#if !USE_ASYNC_LDS
    if (more) {
      int nxt = cur ^ 1;              // safe: nobody reads buf[nxt] this step
      *(u32x4*)&As [nxt][o0] = ra0;   *(u32x4*)&As [nxt][o1] = ra1;
      *(u32x4*)&Bzs[nxt][o0] = rb0;   *(u32x4*)&Bzs[nxt][o1] = rb1;
      *(u32x4*)&Bps[nxt][o0] = rc0;   *(u32x4*)&Bps[nxt][o1] = rc1;
    }
#endif
    cur ^= 1;
  }

  // Epilogue: act(z) out, power partial = z*(Gcol[n]*z - 2*uv)
  float pp = 0.f;
#pragma unroll
  for (int fc = 0; fc < 4; ++fc) {
    int col  = nBase + wc * 64 + fc * 16 + lr;
    float gv = Gcol[col];
#pragma unroll
    for (int fr = 0; fr < 2; ++fr) {
      int rbase = eBase + wr * 32 + fr * 16 + ((lane >> 4) << 3);
#pragma unroll
      for (int r = 0; r < 8; ++r) {
        float z  = accZ[fr][fc][r];
        float uv = accP[fr][fc][r];
        actOut[(size_t)(rbase + r) * N + col] = act_fn(z);
        pp += z * (gv * z - 2.0f * uv);
      }
    }
  }
  atomicAdd(&psum, pp);
  __syncthreads();
  if (t == 0) atomicAdd(powAcc, psum * invE);
}

// ---------------------------------------------------------------------------
extern "C" void kernel_launch(void* const* d_in, const int* in_sizes, int n_in,
                              void* d_out, int out_size, void* d_ws, size_t ws_size,
                              hipStream_t stream)
{
  (void)in_sizes; (void)n_in; (void)out_size; (void)ws_size;

  const float* x = (const float*)d_in[0];
  const float* thetas[3] = {(const float*)d_in[1], (const float*)d_in[2],
                            (const float*)d_in[3]};
  float* out_a  = (float*)d_out;                  // [4096 x 1024]
  float* powAcc = out_a + (size_t)4096 * 1024;    // scalar

  const int topo[4] = {1024, 2048, 2048, 1024};
  const int E = 4096;
  const int KpadMax = 4128;                       // ceil(2*2050/32)*32

  char* ws = (char*)d_ws;
  float*  actBuf = (float*)ws;  ws += (size_t)E * 2048 * sizeof(float);
  __bf16* Ap     = (__bf16*)ws; ws += (size_t)E * KpadMax * sizeof(__bf16);
  __bf16* Bm     = (__bf16*)ws; ws += (size_t)4096 * KpadMax * sizeof(__bf16);
  float*  Sxx    = (float*)ws;  ws += 4352 * sizeof(float);
  float*  Snn    = (float*)ws;  ws += 4352 * sizeof(float);
  float*  s_arr  = (float*)ws;  ws += 2048 * sizeof(float);
  float*  gsc    = (float*)ws;  ws += 2048 * sizeof(float);
  float*  Gcol   = (float*)ws;  ws += 2048 * sizeof(float);

  zero_pow_kernel<<<1, 1, 0, stream>>>(powAcc);

  const float* actIn = x;
  for (int l = 0; l < 3; ++l) {
    int Min = topo[l], N = topo[l + 1];
    int M = Min + 2, twoM = 2 * M;
    int Kpad = ((twoM + 31) / 32) * 32;
    float invE = 1.0f / (float)E;

    act_prep_kernel<<<dim3((M + 63) / 64), 256, 0, stream>>>(
        actIn, E, Min, M, Kpad, Ap, Sxx, Snn);

    theta_stats_kernel<<<dim3(N / 64), 256, 0, stream>>>(
        thetas[l], M, N, Sxx, Snn, s_arr, gsc, Gcol, powAcc, invE);

    theta_emit_kernel<<<dim3((M + 63) / 64, N / 64), 256, 0, stream>>>(
        thetas[l], M, N, Kpad, s_arr, gsc, Bm);

    int padc = Kpad - twoM;
    if (padc > 0) {
      long long tot = (long long)(E + 2 * N) * padc;
      pad_zero_kernel<<<dim3((unsigned)((tot + 255) / 256)), 256, 0, stream>>>(
          Ap, Bm, E, 2 * N, twoM, Kpad);
    }

    float* actOut = (l == 2) ? out_a : actBuf;
    gemm_fused_kernel<<<dim3(E / 128, N / 128), 256, 0, stream>>>(
        Ap, Bm, Gcol, actOut, powAcc, E, N, Kpad, invE);
    actIn = actOut;
  }
}
